// CausalAttention_51840255262800
// MI455X (gfx1250) — matrix operand order repaired and checked by reading, not compiled, not run on hardware
//
#include <hip/hip_runtime.h>
#include <hip/hip_bf16.h>
#include <math.h>

#define B_  2
#define T_  2048
#define C_  2048
#define H_  16
#define D_  128
#define M_  (B_*T_)   // 4096 rows of tokens

#define USE_ASYNC_LDS 1

typedef __attribute__((ext_vector_type(16))) _Float16 v16h;
typedef __attribute__((ext_vector_type(8)))  _Float16 v8h;
typedef __attribute__((ext_vector_type(8)))  float    v8f;

union FragU { v16h v; v8h h[2]; };

// A-matrix 16x32 f16 fragment (ISA 16-bit A layout):
// lanes 0-15: row=lane, halves 0-7 = K0..7, halves 8-15 = K16..23
// lanes 16-31: row=lane-16, halves 0-7 = K8..15, halves 8-15 = K24..31
__device__ __forceinline__ v16h load_frag_a(const _Float16* base, int ld, int lane) {
  const int r = lane & 15, ko = (lane >> 4) * 8;
  const _Float16* p = base + (size_t)r * ld + ko;
  FragU f;
  f.h[0] = *(const v8h*)(p);
  f.h[1] = *(const v8h*)(p + 16);
  return f.v;
}

// B-matrix 32x16 f16 fragment:
// lanes 0-15: col=lane, halves 0-7 = K0..7, 8-15 = K16..23 ; lanes 16-31: col=lane-16, K8..15, K24..31
// `base` points at a [16 cols x K] N-major (transposed) tile.
__device__ __forceinline__ v16h load_frag_b(const _Float16* base, int ld, int lane) {
  const int c = lane & 15, ko = (lane >> 4) * 8;
  const _Float16* p = base + (size_t)c * ld + ko;
  FragU f;
  f.h[0] = *(const v8h*)(p);
  f.h[1] = *(const v8h*)(p + 16);
  return f.v;
}

__device__ __forceinline__ v8f wmma_f16(v16h a, v16h b, v8f c) {
  return __builtin_amdgcn_wmma_f32_16x16x32_f16(false, a, false, b, (short)0, c,
                                                false, false);
}

__device__ __forceinline__ v8f zero_v8f() {
  v8f z;
#pragma unroll
  for (int e = 0; e < 8; ++e) z[e] = 0.f;
  return z;
}

// Async copy 16B/lane from global to LDS (gfx1250 ASYNCcnt path).
__device__ __forceinline__ void async_copy16(const _Float16* g, _Float16* l) {
#if USE_ASYNC_LDS
  const unsigned ldsOff = (unsigned)(size_t)l;  // addr[31:0] = LDS offset
  asm volatile("global_load_async_to_lds_b128 %0, %1, off"
               :: "v"(ldsOff), "v"((unsigned long long)(size_t)g)
               : "memory");
#else
  *(v8h*)l = *(const v8h*)g;
#endif
}
__device__ __forceinline__ void wait_asynccnt_le2() {
#if USE_ASYNC_LDS
  asm volatile("s_wait_asynccnt 0x2" ::: "memory");
#endif
}
__device__ __forceinline__ void wait_asynccnt_0() {
#if USE_ASYNC_LDS
  asm volatile("s_wait_asynccnt 0x0" ::: "memory");
#endif
}

// ---------------- conversion kernels ----------------

__global__ void cvt_f32_to_f16(const float* __restrict__ in,
                               _Float16* __restrict__ out, int n) {
  int i = blockIdx.x * blockDim.x + threadIdx.x;
  if (i < n) out[i] = (_Float16)in[i];
}

// W [rows=K][cols=N] fp32 -> WT [N][K] f16
__global__ void tcvt_f32_to_f16(const float* __restrict__ in,
                                _Float16* __restrict__ outT, int rows, int cols) {
  int i = blockIdx.x * blockDim.x + threadIdx.x;
  if (i < rows * cols) {
    int r = i / cols, c = i % cols;
    outT[(size_t)c * rows + r] = (_Float16)in[i];
  }
}

// ---------------- fused GEMM (+bias, +epilogue) ----------------
// Block: 8 waves, each wave computes a 16(M) x 128(N) strip (full rows so the
// QK L2-norm over D=128 is an intra-wave shuffle reduction).
// B tile (128 cols x 32 k, 8KB) is staged in LDS via async-to-LDS copies,
// double-buffered so step i+1's DMA overlaps step i's WMMAs.
// MODE 0: Q/K  -> bias, L2-normalize per head row, f16 out [B,H,T,D]
// MODE 1: V    -> bias, f16 out transposed [B,H,D,T]
// MODE 2: O    -> bias, fp32 out [M_,C_]
template <int MODE>
__global__ __launch_bounds__(256) void gemm_kernel(
    const _Float16* __restrict__ A,    // [M_,C_] f16 row-major
    const _Float16* __restrict__ BT,   // [C_,C_] f16 stored [N][K]
    const float* __restrict__ bias,    // [C_]
    _Float16* __restrict__ outH,
    float* __restrict__ outF) {
  __shared__ alignas(16) _Float16 Bs[2][128][32];

  const int tid  = threadIdx.x;
  const int lane = tid & 31;
  const int wid  = tid >> 5;
  const int rowBase = blockIdx.y * 128 + wid * 16;
  const int colBase = blockIdx.x * 128;

  // 512 16B-segments per tile; 2 per thread.
  auto issueB = [&](int step) {
#pragma unroll
    for (int t = 0; t < 2; ++t) {
      const int li  = t * 256 + tid;
      const int col = li >> 2;
      const int seg = (li & 3) * 8;  // halves
      const _Float16* g = BT + (size_t)(colBase + col) * C_ + step * 32 + seg;
      async_copy16(g, &Bs[step & 1][col][seg]);
    }
  };

  v8f acc[8];
#pragma unroll
  for (int j = 0; j < 8; ++j) acc[j] = zero_v8f();

  constexpr int NSTEP = C_ / 32;  // 64
  issueB(0);
  for (int i = 0; i < NSTEP; ++i) {
    v16h a = load_frag_a(A + (size_t)rowBase * C_ + i * 32, C_, lane);
    if (i + 1 < NSTEP) { issueB(i + 1); wait_asynccnt_le2(); }
    else               { wait_asynccnt_0(); }
    __syncthreads();  // all waves' step-i tile landed in LDS
#pragma unroll
    for (int j = 0; j < 8; ++j) {
      v16h b = load_frag_b(&Bs[i & 1][j * 16][0], 32, lane);
      acc[j] = wmma_f16(a, b, acc[j]);
    }
    __syncthreads();  // nobody still reads this buffer when it is re-filled
  }

  const int hs = lane >> 4;  // half of the wave
  const int n  = lane & 15;  // column-within-tile

#pragma unroll
  for (int j = 0; j < 8; ++j) {
    float bc = bias[colBase + j * 16 + n];
#pragma unroll
    for (int i = 0; i < 8; ++i) acc[j][i] += bc;
  }

  if (MODE == 0) {
#pragma unroll
    for (int i = 0; i < 8; ++i) {
      float ss = 0.f;
#pragma unroll
      for (int j = 0; j < 8; ++j) ss += acc[j][i] * acc[j][i];
      ss += __shfl_xor(ss, 1, 32);
      ss += __shfl_xor(ss, 2, 32);
      ss += __shfl_xor(ss, 4, 32);
      ss += __shfl_xor(ss, 8, 32);
      const float rn = rsqrtf(ss);
      const int row = rowBase + i + 8 * hs;
      const int b = row >> 11, t = row & (T_ - 1);
      const int h = colBase >> 7;
#pragma unroll
      for (int j = 0; j < 8; ++j) {
        const int d = j * 16 + n;
        outH[(((size_t)b * H_ + h) * T_ + t) * D_ + d] = (_Float16)(acc[j][i] * rn);
      }
    }
  } else if (MODE == 1) {
#pragma unroll
    for (int i = 0; i < 8; ++i) {
      const int row = rowBase + i + 8 * hs;
      const int b = row >> 11, t = row & (T_ - 1);
      const int h = colBase >> 7;
#pragma unroll
      for (int j = 0; j < 8; ++j) {
        const int d = j * 16 + n;
        outH[(((size_t)b * H_ + h) * D_ + d) * T_ + t] = (_Float16)acc[j][i];
      }
    }
  } else {
#pragma unroll
    for (int i = 0; i < 8; ++i) {
      const int row = rowBase + i + 8 * hs;
#pragma unroll
      for (int j = 0; j < 8; ++j)
        outF[(size_t)row * C_ + colBase + j * 16 + n] = acc[j][i];
    }
  }
}

// ---------------- flash attention ----------------
// grid(B*H, T/128), 8 waves/block, each wave = 16 query rows, full D=128.
__global__ __launch_bounds__(256) void attn_kernel(
    const _Float16* __restrict__ Q,   // [B,H,T,D] f16, L2-normalized
    const _Float16* __restrict__ Km,  // [B,H,T,D] f16, L2-normalized
    const _Float16* __restrict__ VT,  // [B,H,D,T] f16
    const float* __restrict__ scale,  // [H]
    _Float16* __restrict__ AO) {      // [B,T,H,D] == [M_,C_] f16
  __shared__ alignas(16) _Float16 Plds[8][16][32];  // per-wave P tile relayout

  const int lane = threadIdx.x & 31;
  const int wid  = threadIdx.x >> 5;
  const int bh = blockIdx.x;
  const int b = bh >> 4, h = bh & 15;
  const int q0 = blockIdx.y * 128 + wid * 16;

  const _Float16* Qp = Q  + (size_t)bh * T_ * D_;
  const _Float16* Kp = Km + (size_t)bh * T_ * D_;
  const _Float16* Vp = VT + (size_t)bh * D_ * T_;

  v16h qf[4];
#pragma unroll
  for (int dc = 0; dc < 4; ++dc)
    qf[dc] = load_frag_a(Qp + (size_t)q0 * D_ + dc * 32, D_, lane);

  const float sc = scale[h];
  const float slope = exp2f(-0.5f * (float)(h + 1));  // H=16 power-of-two slopes
  const int hs = lane >> 4, n = lane & 15;
  const int qrowBase = q0 + 8 * hs;

  v8f o[8];
#pragma unroll
  for (int j = 0; j < 8; ++j) o[j] = zero_v8f();
  float mrow[8], lrow[8];
#pragma unroll
  for (int i = 0; i < 8; ++i) { mrow[i] = -1e30f; lrow[i] = 0.f; }

  for (int kb = 0; kb < q0 + 16; kb += 32) {
    // hint next key block while this one computes (global_prefetch_b8)
    if (kb + 32 < q0 + 16) {
      __builtin_prefetch(Kp + (size_t)(kb + 32 + lane) * D_, 0, 3);
      __builtin_prefetch(Vp + (size_t)(lane * 4) * T_ + kb + 32, 0, 3);
    }
    // S = scale * q @ k^T  (two 16x16 key tiles, K-dim = D in 4 chunks)
    v8f s0 = zero_v8f(), s1 = zero_v8f();
#pragma unroll
    for (int dc = 0; dc < 4; ++dc) {
      v16h k0f = load_frag_b(Kp + (size_t)kb * D_ + dc * 32, D_, lane);
      v16h k1f = load_frag_b(Kp + (size_t)(kb + 16) * D_ + dc * 32, D_, lane);
      s0 = wmma_f16(qf[dc], k0f, s0);
      s1 = wmma_f16(qf[dc], k1f, s1);
    }
    const int j0 = kb + n, j1 = kb + 16 + n;
    const float al0 = slope * (float)(j0 - (T_ - 1));  // ALiBi: slope*-(T-1-j)
    const float al1 = slope * (float)(j1 - (T_ - 1));

    float alpha[8];
#pragma unroll
    for (int i = 0; i < 8; ++i) {
      const int qr = qrowBase + i;
      const bool u0 = (j0 <= qr), u1 = (j1 <= qr);  // causal
      const float e0 = u0 ? (s0[i] * sc + al0) : -1e30f;
      const float e1 = u1 ? (s1[i] * sc + al1) : -1e30f;
      float mx = fmaxf(e0, e1);
      mx = fmaxf(mx, __shfl_xor(mx, 1, 32));
      mx = fmaxf(mx, __shfl_xor(mx, 2, 32));
      mx = fmaxf(mx, __shfl_xor(mx, 4, 32));
      mx = fmaxf(mx, __shfl_xor(mx, 8, 32));
      const float nm = fmaxf(mrow[i], mx);
      alpha[i] = __expf(mrow[i] - nm);
      mrow[i] = nm;
      const float p0 = u0 ? __expf(e0 - nm) : 0.f;
      const float p1 = u1 ? __expf(e1 - nm) : 0.f;
      float rs = p0 + p1;
      rs += __shfl_xor(rs, 1, 32);
      rs += __shfl_xor(rs, 2, 32);
      rs += __shfl_xor(rs, 4, 32);
      rs += __shfl_xor(rs, 8, 32);
      lrow[i] = lrow[i] * alpha[i] + rs;
      Plds[wid][i + 8 * hs][n]      = (_Float16)p0;   // C-layout -> row-major LDS
      Plds[wid][i + 8 * hs][16 + n] = (_Float16)p1;
    }
#pragma unroll
    for (int j = 0; j < 8; ++j)
#pragma unroll
      for (int i = 0; i < 8; ++i) o[j][i] *= alpha[i];

    // wave-private LDS round-trip: P (C-layout fp32) -> A-fragment f16
    v16h pf = load_frag_a(&Plds[wid][0][0], 32, lane);
#pragma unroll
    for (int j = 0; j < 8; ++j) {
      v16h vf = load_frag_b(Vp + (size_t)(j * 16) * T_ + kb, T_, lane);
      o[j] = wmma_f16(pf, vf, o[j]);
    }
  }

#pragma unroll
  for (int i = 0; i < 8; ++i) {
    const float inv = 1.f / lrow[i];
    const size_t outRow = (size_t)b * T_ + (qrowBase + i);
#pragma unroll
    for (int j = 0; j < 8; ++j) {
      const int c = h * D_ + j * 16 + n;
      AO[outRow * C_ + c] = (_Float16)(o[j][i] * inv);
    }
  }
}

// ---------------- launch ----------------

extern "C" void kernel_launch(void* const* d_in, const int* in_sizes, int n_in,
                              void* d_out, int out_size, void* d_ws, size_t ws_size,
                              hipStream_t stream) {
  (void)in_sizes; (void)n_in; (void)out_size; (void)ws_size;
  const float* x     = (const float*)d_in[0];
  const float* Wq    = (const float*)d_in[1];
  const float* bq    = (const float*)d_in[2];
  const float* Wk    = (const float*)d_in[3];
  const float* bk    = (const float*)d_in[4];
  const float* Wv    = (const float*)d_in[5];
  const float* bv    = (const float*)d_in[6];
  const float* Wo    = (const float*)d_in[7];
  const float* bo    = (const float*)d_in[8];
  const float* scale = (const float*)d_in[9];
  float* out = (float*)d_out;

  char* w = (char*)d_ws;
  _Float16* xh  = (_Float16*)w; w += (size_t)M_ * C_ * 2;   // 16 MB
  _Float16* WqT = (_Float16*)w; w += (size_t)C_ * C_ * 2;   //  8 MB each
  _Float16* WkT = (_Float16*)w; w += (size_t)C_ * C_ * 2;
  _Float16* WvT = (_Float16*)w; w += (size_t)C_ * C_ * 2;
  _Float16* WoT = (_Float16*)w; w += (size_t)C_ * C_ * 2;
  _Float16* Qb  = (_Float16*)w; w += (size_t)M_ * C_ * 2;   // [B,H,T,D]
  _Float16* Kb  = (_Float16*)w; w += (size_t)M_ * C_ * 2;   // [B,H,T,D]
  _Float16* VTb = (_Float16*)w; w += (size_t)M_ * C_ * 2;   // [B,H,D,T]
  _Float16* AOb = xh;  // xh is dead after the V GEMM; reuse for attention out

  const int thr = 256;
  cvt_f32_to_f16<<<(M_ * C_ + thr - 1) / thr, thr, 0, stream>>>(x, xh, M_ * C_);
  const int nw = C_ * C_;
  tcvt_f32_to_f16<<<(nw + thr - 1) / thr, thr, 0, stream>>>(Wq, WqT, C_, C_);
  tcvt_f32_to_f16<<<(nw + thr - 1) / thr, thr, 0, stream>>>(Wk, WkT, C_, C_);
  tcvt_f32_to_f16<<<(nw + thr - 1) / thr, thr, 0, stream>>>(Wv, WvT, C_, C_);
  tcvt_f32_to_f16<<<(nw + thr - 1) / thr, thr, 0, stream>>>(Wo, WoT, C_, C_);

  dim3 gg(C_ / 128, M_ / 128);
  gemm_kernel<0><<<gg, thr, 0, stream>>>(xh, WqT, bq, Qb,  nullptr);
  gemm_kernel<0><<<gg, thr, 0, stream>>>(xh, WkT, bk, Kb,  nullptr);
  gemm_kernel<1><<<gg, thr, 0, stream>>>(xh, WvT, bv, VTb, nullptr);

  attn_kernel<<<dim3(B_ * H_, T_ / 128), thr, 0, stream>>>(Qb, Kb, VTb, scale, AOb);

  gemm_kernel<2><<<gg, thr, 0, stream>>>(AOb, WoT, bo, nullptr, out);
}
